// BinaryTwoDimRNNModel_34016140984582
// MI455X (gfx1250) — matrix operand order repaired and checked by reading, not compiled, not run on hardware
//
#include <hip/hip_runtime.h>
#include <hip/hip_bf16.h>
#include <math.h>

#define RB 32
#define RT 256
#define RI 128
#define RH 4096

#define WPB 8   // waves per block; each wave owns one (M-tile, N-tile) pair

typedef __attribute__((ext_vector_type(16))) __bf16 v16bf;
typedef __attribute__((ext_vector_type(8)))  float  v8f;
typedef __hip_bfloat16 bf16_t;

union FragBF {
  v16bf v;
  uint4 q[2];
};

// A-matrix fragment (16x32 bf16): lane<16 -> row=lane, K = kb+{0..7} and kb+16+{0..7}
//                                 lane>=16 -> row=lane-16, K = kb+8+{0..7} and kb+24+{0..7}
__device__ __forceinline__ v16bf load_frag_a(const bf16_t* __restrict__ base, int ld,
                                             int row, int kb, int half) {
  const bf16_t* p = base + (size_t)row * ld + kb + half * 8;
  FragBF f;
  f.q[0] = *(const uint4*)(p);
  f.q[1] = *(const uint4*)(p + 16);
  return f.v;
}

// B-matrix fragment (32x16, B[k,n] = W[n,k]): lane column n = lm, K = kb+{0..7} and kb+16+{0..7}
// (lanes 0-15) or kb+8+{0..7} and kb+24+{0..7} (lanes 16-31); matches the A fragment.
__device__ __forceinline__ v16bf load_frag_b(const bf16_t* __restrict__ w, int ldw,
                                             int col, int kb, int half) {
  const bf16_t* p = w + (size_t)col * ldw + kb + half * 8;
  FragBF f;
  f.q[0] = *(const uint4*)(p);
  f.q[1] = *(const uint4*)(p + 16);
  return f.v;
}

// One RNN time step: Hnext = tanh(A0[32,K0] @ W0[H,K0]^T + Hprev[32,H] @ Whh[H,H]^T + b_ih + b_hh)
// One 16x16 output tile per wave (512 waves/step); two independent WMMA accumulator
// chains per wave (even/odd K chunks) to keep the XDL pipe fed despite RAW latency.
__global__ __launch_bounds__(WPB * 32)
void rnn_step_kernel(const bf16_t* __restrict__ A0, int lda0, int K0,
                     const bf16_t* __restrict__ W0,
                     const bf16_t* __restrict__ Hprev,
                     const bf16_t* __restrict__ Whh,
                     const float* __restrict__ b_ih,
                     const float* __restrict__ b_hh,
                     float* __restrict__ Hnext_f,
                     bf16_t* __restrict__ Hnext_bf,
                     bf16_t* __restrict__ Yout, int ldy) {
  const int lane = threadIdx.x & 31;
  const int wave = threadIdx.x >> 5;
  const int half = lane >> 4;
  const int lm   = lane & 15;

  const int wgid = blockIdx.x * WPB + wave;  // 0 .. 511
  const int mt   = wgid & 1;                 // M-tile (rows mt*16 .. mt*16+15)
  const int nt   = wgid >> 1;                // N-tile (cols nt*16 .. nt*16+15)
  const int mrow = mt * 16 + lm;             // this lane's A row (both halves)
  const int ncol = nt * 16;

  v8f acc0, acc1;
#pragma unroll
  for (int e = 0; e < 8; ++e) { acc0[e] = 0.0f; acc1[e] = 0.0f; }

  // ---- input-projection contribution: A0[32,K0] x W0^T (K0 % 64 == 0) ----
  for (int kb = 0; kb < K0; kb += 64) {
    v16bf a0 = load_frag_a(A0, lda0, mrow, kb,      half);
    v16bf b0 = load_frag_b(W0, K0, ncol + lm, kb,   half);
    v16bf a1 = load_frag_a(A0, lda0, mrow, kb + 32, half);
    v16bf b1 = load_frag_b(W0, K0, ncol + lm, kb + 32, half);
    acc0 = __builtin_amdgcn_wmma_f32_16x16x32_bf16(false, a0, false, b0,
                                                   (short)0, acc0, false, false);
    acc1 = __builtin_amdgcn_wmma_f32_16x16x32_bf16(false, a1, false, b1,
                                                   (short)0, acc1, false, false);
  }

  // ---- recurrent contribution: Hprev[32,H] x Whh^T ----
  for (int kb = 0; kb < RH; kb += 64) {
    v16bf a0 = load_frag_a(Hprev, RH, mrow, kb,      half);
    v16bf b0 = load_frag_b(Whh, RH, ncol + lm, kb,   half);
    v16bf a1 = load_frag_a(Hprev, RH, mrow, kb + 32, half);
    v16bf b1 = load_frag_b(Whh, RH, ncol + lm, kb + 32, half);
    // pull next weight chunk toward the caches (global_prefetch_b8)
    __builtin_prefetch(Whh + (size_t)(ncol + lm) * RH + kb + 256, 0, 0);
    acc0 = __builtin_amdgcn_wmma_f32_16x16x32_bf16(false, a0, false, b0,
                                                   (short)0, acc0, false, false);
    acc1 = __builtin_amdgcn_wmma_f32_16x16x32_bf16(false, a1, false, b1,
                                                   (short)0, acc1, false, false);
  }

  // ---- epilogue: merge chains, bias + tanh, write fp32 state, bf16 state, y0 slice ----
  const int n = ncol + lm;
  const float bias = b_ih[n] + b_hh[n];
#pragma unroll
  for (int r = 0; r < 8; ++r) {
    const int m = mt * 16 + half * 8 + r;    // C/D layout: VGPR r -> row r (+8 for hi lanes)
    const float v = tanhf(acc0[r] + acc1[r] + bias);
    Hnext_f[(size_t)m * RH + n] = v;
    const bf16_t vb = __float2bfloat16(v);
    Hnext_bf[(size_t)m * RH + n] = vb;
    if (Yout) Yout[(size_t)m * ldy + n] = vb;
  }
}

__global__ void f32_to_bf16_kernel(const float* __restrict__ s, bf16_t* __restrict__ d, int n) {
  int i = blockIdx.x * blockDim.x + threadIdx.x;
  if (i < n) d[i] = __float2bfloat16(s[i]);
}

__global__ void zero_bf16_kernel(bf16_t* __restrict__ d, int n) {
  int i = blockIdx.x * blockDim.x + threadIdx.x;
  if (i < n) d[i] = __float2bfloat16(0.0f);
}

// conv 5x5 pad(1,1) on 64x64 -> 62x62, + bias, ReLU
__global__ void conv_relu_kernel(const float* __restrict__ h0, const float* __restrict__ h1,
                                 const float* __restrict__ cw, const float* __restrict__ cb,
                                 float* __restrict__ outc) {
  int idx = blockIdx.x * blockDim.x + threadIdx.x;
  if (idx >= 64 * 62 * 62) return;
  int img = idx / (62 * 62);
  int rem = idx % (62 * 62);
  int y = rem / 62, x = rem % 62;
  const float* im = (img < 32) ? (h0 + (size_t)img * RH) : (h1 + (size_t)(img - 32) * RH);
  float s = cb[0];
  for (int ky = 0; ky < 5; ++ky) {
    int iy = y - 1 + ky;
    if (iy < 0 || iy > 63) continue;
    for (int kx = 0; kx < 5; ++kx) {
      int ix = x - 1 + kx;
      if (ix < 0 || ix > 63) continue;
      s += im[iy * 64 + ix] * cw[ky * 5 + kx];
    }
  }
  outc[idx] = fmaxf(s, 0.0f);
}

// 3x3 stride-2 VALID maxpool on 62x62 -> 30x30
__global__ void maxpool_kernel(const float* __restrict__ c, float* __restrict__ p) {
  int idx = blockIdx.x * blockDim.x + threadIdx.x;
  if (idx >= 64 * 30 * 30) return;
  int img = idx / 900;
  int rem = idx % 900;
  int py = rem / 30, px = rem % 30;
  const float* ci = c + (size_t)img * 62 * 62;
  float m = -INFINITY;
  for (int ky = 0; ky < 3; ++ky)
    for (int kx = 0; kx < 3; ++kx)
      m = fmaxf(m, ci[(2 * py + ky) * 62 + (2 * px + kx)]);
  p[idx] = m;
}

// half-pixel (align_corners=False) bilinear 30x30 -> 64x64, then sigmoid
__global__ void resize_sigmoid_kernel(const float* __restrict__ p, float* __restrict__ o) {
  int idx = blockIdx.x * blockDim.x + threadIdx.x;
  if (idx >= 64 * 64 * 64) return;
  int img = idx >> 12;
  int rem = idx & 4095;
  int oy = rem >> 6, ox = rem & 63;
  const float sc = 30.0f / 64.0f;
  float sy = (oy + 0.5f) * sc - 0.5f;
  float sx = (ox + 0.5f) * sc - 0.5f;
  int y0 = (int)floorf(sy), x0 = (int)floorf(sx);
  float fy = sy - (float)y0, fx = sx - (float)x0;
  int y1 = y0 + 1, x1 = x0 + 1;
  y0 = y0 < 0 ? 0 : (y0 > 29 ? 29 : y0);
  y1 = y1 < 0 ? 0 : (y1 > 29 ? 29 : y1);
  x0 = x0 < 0 ? 0 : (x0 > 29 ? 29 : x0);
  x1 = x1 < 0 ? 0 : (x1 > 29 ? 29 : x1);
  const float* pi = p + (size_t)img * 900;
  float v00 = pi[y0 * 30 + x0], v01 = pi[y0 * 30 + x1];
  float v10 = pi[y1 * 30 + x0], v11 = pi[y1 * 30 + x1];
  float v = (1.0f - fy) * ((1.0f - fx) * v00 + fx * v01) +
            fy * ((1.0f - fx) * v10 + fx * v11);
  o[idx] = 1.0f / (1.0f + expf(-v));
}

extern "C" void kernel_launch(void* const* d_in, const int* in_sizes, int n_in,
                              void* d_out, int out_size, void* d_ws, size_t ws_size,
                              hipStream_t stream) {
  (void)in_sizes; (void)n_in; (void)out_size; (void)ws_size;
  const float* x      = (const float*)d_in[0];
  const float* w_ih0  = (const float*)d_in[1];
  const float* w_hh0  = (const float*)d_in[2];
  const float* b_ih0  = (const float*)d_in[3];
  const float* b_hh0  = (const float*)d_in[4];
  const float* w_ih1  = (const float*)d_in[5];
  const float* w_hh1  = (const float*)d_in[6];
  const float* b_ih1  = (const float*)d_in[7];
  const float* b_hh1  = (const float*)d_in[8];
  const float* conv_w = (const float*)d_in[9];
  const float* conv_b = (const float*)d_in[10];
  float* out = (float*)d_out;

  char* ws = (char*)d_ws;
  size_t off = 0;
  auto alloc = [&](size_t bytes) -> void* {
    off = (off + 255) & ~(size_t)255;
    void* p = ws + off;
    off += bytes;
    return p;
  };

  bf16_t* x_bf    = (bf16_t*)alloc((size_t)RB * RT * RI * 2);
  bf16_t* wih0_bf = (bf16_t*)alloc((size_t)RH * RI * 2);
  bf16_t* whh0_bf = (bf16_t*)alloc((size_t)RH * RH * 2);
  bf16_t* wih1_bf = (bf16_t*)alloc((size_t)RH * RH * 2);
  bf16_t* whh1_bf = (bf16_t*)alloc((size_t)RH * RH * 2);
  bf16_t* y0_bf   = (bf16_t*)alloc((size_t)RB * RT * RH * 2);
  bf16_t* h0bf[2] = {(bf16_t*)alloc((size_t)RB * RH * 2), (bf16_t*)alloc((size_t)RB * RH * 2)};
  bf16_t* h1bf[2] = {(bf16_t*)alloc((size_t)RB * RH * 2), (bf16_t*)alloc((size_t)RB * RH * 2)};
  float*  h0f[2]  = {(float*)alloc((size_t)RB * RH * 4), (float*)alloc((size_t)RB * RH * 4)};
  float*  h1f[2]  = {(float*)alloc((size_t)RB * RH * 4), (float*)alloc((size_t)RB * RH * 4)};
  float*  c1 = (float*)alloc((size_t)64 * 62 * 62 * 4);
  float*  pl = (float*)alloc((size_t)64 * 30 * 30 * 4);

  auto cvt = [&](const float* s, bf16_t* d, int n) {
    f32_to_bf16_kernel<<<(n + 255) / 256, 256, 0, stream>>>(s, d, n);
  };
  cvt(x,     x_bf,    RB * RT * RI);
  cvt(w_ih0, wih0_bf, RH * RI);
  cvt(w_hh0, whh0_bf, RH * RH);
  cvt(w_ih1, wih1_bf, RH * RH);
  cvt(w_hh1, whh1_bf, RH * RH);

  zero_bf16_kernel<<<(RB * RH + 255) / 256, 256, 0, stream>>>(h0bf[0], RB * RH);
  zero_bf16_kernel<<<(RB * RH + 255) / 256, 256, 0, stream>>>(h1bf[0], RB * RH);

  // 512 output tiles (2 M-tiles x 256 N-tiles), one per wave, 8 waves/block -> 64 blocks
  const int grid = (2 * (RH / 16)) / WPB;

  // layer 0: fused input-GEMM (K=128) + recurrence; emit y0[:,t,:] as bf16
  for (int t = 0; t < RT; ++t) {
    rnn_step_kernel<<<grid, WPB * 32, 0, stream>>>(
        x_bf + (size_t)t * RI, RT * RI, RI, wih0_bf,
        h0bf[t & 1], whh0_bf, b_ih0, b_hh0,
        h0f[(t + 1) & 1], h0bf[(t + 1) & 1],
        y0_bf + (size_t)t * RH, RT * RH);
  }
  // layer 1: fused input-GEMM (K=4096 from y0) + recurrence
  for (int t = 0; t < RT; ++t) {
    rnn_step_kernel<<<grid, WPB * 32, 0, stream>>>(
        y0_bf + (size_t)t * RH, RT * RH, RH, wih1_bf,
        h1bf[t & 1], whh1_bf, b_ih1, b_hh1,
        h1f[(t + 1) & 1], h1bf[(t + 1) & 1],
        nullptr, 0);
  }

  // final hiddens land in buffer 0 (t=255 writes (255+1)&1 == 0)
  conv_relu_kernel<<<(64 * 62 * 62 + 255) / 256, 256, 0, stream>>>(h0f[0], h1f[0], conv_w, conv_b, c1);
  maxpool_kernel<<<(64 * 30 * 30 + 255) / 256, 256, 0, stream>>>(c1, pl);
  resize_sigmoid_kernel<<<(64 * 64 * 64 + 255) / 256, 256, 0, stream>>>(pl, out);
}
